// GTv1GraphNN_79293686218849
// MI455X (gfx1250) — matrix operand order repaired and checked by reading, not compiled, not run on hardware
//
#include <hip/hip_runtime.h>
#include <hip/hip_bf16.h>
#include <math.h>
#include <stdint.h>

typedef __bf16 bf16_t;
typedef bf16_t v16bf __attribute__((ext_vector_type(16)));
typedef float  v8f   __attribute__((ext_vector_type(8)));

#define D_IN    126
#define HSTRIDE 2048

// ---------------------------------------------------------------------------
// CDNA5 async global->LDS copy (16B per lane) + ASYNCcnt wait
// ---------------------------------------------------------------------------
__device__ __forceinline__ void async_copy_b128(unsigned lds_off, const void* gaddr) {
  asm volatile("global_load_async_to_lds_b128 %0, %1, off"
               :: "v"(lds_off), "v"((unsigned long long)(uintptr_t)gaddr)
               : "memory");
}

#if __has_builtin(__builtin_amdgcn_s_wait_asynccnt)
#define WAIT_ASYNCCNT(n) __builtin_amdgcn_s_wait_asynccnt(n)
#else
#define WAIT_ASYNCCNT(n) asm volatile("s_wait_asynccnt %0" :: "n"(n) : "memory")
#endif

// ---------------------------------------------------------------------------
// small elementwise / graph kernels
// ---------------------------------------------------------------------------
__device__ __forceinline__ float silu_f(float x) {
  return x / (1.0f + __expf(-x));
}

__global__ void fill_f32(float* p, float v, int n) {
  int i = blockIdx.x * blockDim.x + threadIdx.x;
  if (i < n) p[i] = v;
}

__global__ void deg_accum(const long long* __restrict__ dst, float* deg, int E) {
  int e = blockIdx.x * blockDim.x + threadIdx.x;
  if (e < E) atomicAdd(&deg[(int)dst[e]], 1.0f);
}

__global__ void rsqrt_inplace(float* p, int n) {
  int i = blockIdx.x * blockDim.x + threadIdx.x;
  if (i < n) p[i] = rsqrtf(p[i]);
}

// x (f32, [n, fin]) -> xb (bf16, [n, kp]) zero-padded in K
__global__ void convert_x_bf16(const float* __restrict__ x, bf16_t* __restrict__ xb,
                               int n, int fin, int kp) {
  long long idx = (long long)blockIdx.x * blockDim.x + threadIdx.x;
  long long tot = (long long)n * kp;
  if (idx >= tot) return;
  int i = (int)(idx / kp), k = (int)(idx % kp);
  float v = (k < fin) ? x[(long long)i * fin + k] : 0.0f;
  xb[idx] = (bf16_t)v;
}

// W (f32, [fin, fout]) -> wbt (bf16, [np, kp]) transposed, zero-padded
__global__ void convert_w_bf16_t(const float* __restrict__ W, bf16_t* __restrict__ wbt,
                                 int fin, int fout, int kp, int np) {
  long long idx = (long long)blockIdx.x * blockDim.x + threadIdx.x;
  long long tot = (long long)np * kp;
  if (idx >= tot) return;
  int nn = (int)(idx / kp), k = (int)(idx % kp);
  float v = (k < fin && nn < fout) ? W[(long long)k * fout + nn] : 0.0f;
  wbt[idx] = (bf16_t)v;
}

// agg[i,f] = h[i,f] * dinv[i]^2 + b[f]   (self-loop term + bias, initializes agg)
__global__ void self_bias(const float* __restrict__ h, const float* __restrict__ dinv,
                          const float* __restrict__ b, float* __restrict__ agg,
                          int n, int F, int astride) {
  long long idx = (long long)blockIdx.x * blockDim.x + threadIdx.x;
  long long tot = (long long)n * F;
  if (idx >= tot) return;
  int i = (int)(idx / F), f = (int)(idx % F);
  float di = dinv[i];
  agg[(long long)i * astride + f] = h[(long long)i * HSTRIDE + f] * di * di + b[f];
}

// agg[dst,f] += h[src,f] * dinv[src]*dinv[dst]   (one block per edge)
__global__ void scatter_edges(const float* __restrict__ h, float* __restrict__ agg,
                              const long long* __restrict__ src,
                              const long long* __restrict__ dst,
                              const float* __restrict__ dinv,
                              int E, int F, int astride) {
  int e = blockIdx.x;
  if (e >= E) return;
  int s = (int)src[e], d = (int)dst[e];
  float nrm = dinv[s] * dinv[d];
  const float* hs = h + (long long)s * HSTRIDE;
  float* ad = agg + (long long)d * astride;
  for (int f = threadIdx.x; f < F; f += blockDim.x)
    atomicAdd(&ad[f], hs[f] * nrm);
}

// xb[i,f] = bf16(silu(agg[i,f]))   (feeds next layer's GEMM A operand)
__global__ void silu_to_bf16(const float* __restrict__ agg, bf16_t* __restrict__ xb,
                             int n, int F) {
  long long idx = (long long)blockIdx.x * blockDim.x + threadIdx.x;
  long long tot = (long long)n * F;
  if (idx >= tot) return;
  int i = (int)(idx / F), f = (int)(idx % F);
  xb[(long long)i * F + f] = (bf16_t)silu_f(agg[(long long)i * HSTRIDE + f]);
}

// ---------------------------------------------------------------------------
// bf16 WMMA GEMM: C[M, cstride] = A[M, Kp] * Bt[Np, Kp]^T  (f32 accumulate)
// block = 256 threads = 8 waves; block tile 128x128; wave tile 32x64
// double-buffered LDS fed by global_load_async_to_lds_b128 (ASYNCcnt)
// ---------------------------------------------------------------------------
#define BM 128
#define BN 128
#define BK 32
#define LDK 40   // LDS row stride in bf16 (80B: 16B-aligned, conflict-free)

__global__ __launch_bounds__(256)
void gemm_bf16_wmma(const bf16_t* __restrict__ A,
                    const bf16_t* __restrict__ Bt,
                    float* __restrict__ C,
                    int M, int Kp, int cstride) {
  __shared__ __align__(16) bf16_t As[2][BM * LDK];
  __shared__ __align__(16) bf16_t Bs[2][BN * LDK];

  const int tid  = threadIdx.x;
  const int lane = tid & 31;
  const int wid  = tid >> 5;
  const int wm   = wid & 3;       // 4 waves along M -> wm*32
  const int wn   = wid >> 2;      // 2 waves along N -> wn*64
  const int half = lane >> 4;
  const int l16  = lane & 15;

  const int m0 = blockIdx.x * BM;
  const int n0 = blockIdx.y * BN;

  // staging coords: each thread owns (r0, c0) and (r0+64, c0) 16B chunks
  const int r0 = tid >> 2;
  const int c0 = tid & 3;

  v8f acc[2][4] = {};

  // issue one 128x32 A tile + 128x32 B tile into LDS buffer `buf`
  // (4 async instructions per wave -> ASYNCcnt +4)
  auto issue_tile = [&](int buf, int k0) {
#pragma unroll
    for (int it = 0; it < 2; ++it) {
      int r  = r0 + it * 64;
      int gm = m0 + r;
      gm = gm < M ? gm : M - 1;   // clamp: rows >= M never stored, branch-free
      unsigned lds = (unsigned)(uintptr_t)&As[buf][r * LDK + c0 * 8];
      async_copy_b128(lds, A + (long long)gm * Kp + k0 + c0 * 8);
    }
#pragma unroll
    for (int it = 0; it < 2; ++it) {
      int r = r0 + it * 64;
      unsigned lds = (unsigned)(uintptr_t)&Bs[buf][r * LDK + c0 * 8];
      async_copy_b128(lds, Bt + (long long)(n0 + r) * Kp + k0 + c0 * 8);
    }
  };

  const int nK = Kp / BK;
  issue_tile(0, 0);

  for (int kt = 0; kt < nK; ++kt) {
    const int cur = kt & 1;
    if (kt + 1 < nK) {
      issue_tile(cur ^ 1, (kt + 1) * BK);  // prefetch next tile into other buffer
      WAIT_ASYNCCNT(4);                    // in-order: current buffer's 4 are done
    } else {
      WAIT_ASYNCCNT(0);
    }
    __syncthreads();                       // all waves' current tile in LDS

    // ---- build fragments per ISA VGPR layouts
    union Frag { uint4 u[2]; v16bf v; };
    Frag a[2], b[4];
#pragma unroll
    for (int tm = 0; tm < 2; ++tm) {
      const bf16_t* base = &As[cur][(wm * 32 + tm * 16 + l16) * LDK];
      // A 16x32: half 0 -> K {0..7,16..23}; half 1 -> K {8..15,24..31}
      a[tm].u[0] = *(const uint4*)(base + half * 8);
      a[tm].u[1] = *(const uint4*)(base + half * 8 + 16);
    }
#pragma unroll
    for (int tn = 0; tn < 4; ++tn) {
      const bf16_t* base = &Bs[cur][(wn * 64 + tn * 16 + l16) * LDK];
      // B 32x16: half 0 -> K {0..7,16..23}; half 1 -> K {8..15,24..31}
      b[tn].u[0] = *(const uint4*)(base + half * 8);
      b[tn].u[1] = *(const uint4*)(base + half * 8 + 16);
    }

#pragma unroll
    for (int tm = 0; tm < 2; ++tm)
#pragma unroll
      for (int tn = 0; tn < 4; ++tn)
        acc[tm][tn] = __builtin_amdgcn_wmma_f32_16x16x32_bf16(
            false, a[tm].v, false, b[tn].v, (short)0, acc[tm][tn], false, false);

    __syncthreads();                       // tile fully consumed before overwrite
  }

  // ---- store C: VGPR r -> M = r + 8*half, N = l16
#pragma unroll
  for (int tm = 0; tm < 2; ++tm)
#pragma unroll
    for (int tn = 0; tn < 4; ++tn)
#pragma unroll
      for (int r = 0; r < 8; ++r) {
        int gm = m0 + wm * 32 + tm * 16 + half * 8 + r;
        int gn = n0 + wn * 64 + tn * 16 + l16;
        if (gm < M) C[(long long)gm * cstride + gn] = acc[tm][tn][r];
      }
}

// ---------------------------------------------------------------------------
// driver
// ---------------------------------------------------------------------------
extern "C" void kernel_launch(void* const* d_in, const int* in_sizes, int n_in,
                              void* d_out, int out_size, void* d_ws, size_t ws_size,
                              hipStream_t stream) {
  (void)n_in; (void)out_size; (void)ws_size;
  const float*     x   = (const float*)d_in[0];
  const long long* ei  = (const long long*)d_in[1];
  const float*     W1  = (const float*)d_in[2];
  const float*     b1  = (const float*)d_in[3];
  const float*     W2  = (const float*)d_in[4];
  const float*     b2  = (const float*)d_in[5];
  const float*     W3  = (const float*)d_in[6];
  const float*     b3  = (const float*)d_in[7];
  float*           out = (float*)d_out;

  const int N = in_sizes[0] / D_IN;     // 50000
  const int E = in_sizes[1] / 2;        // 1.6M
  const long long* src = ei;
  const long long* dst = ei + E;

  // workspace carve (256B aligned)
  char* w = (char*)d_ws;
  auto carve = [&](size_t bytes) {
    void* p = (void*)w;
    w += (bytes + 255) & ~(size_t)255;
    return p;
  };
  float*  dinv = (float*)carve(sizeof(float) * (size_t)N);               // deg -> dinv
  float*  h    = (float*)carve(sizeof(float) * (size_t)N * HSTRIDE);     // GEMM out
  float*  agg  = (float*)carve(sizeof(float) * (size_t)N * HSTRIDE);     // aggregate
  bf16_t* xb   = (bf16_t*)carve(sizeof(bf16_t) * (size_t)N * HSTRIDE);   // bf16 act
  bf16_t* wbt  = (bf16_t*)carve(sizeof(bf16_t) * (size_t)2048 * 2048);   // bf16 W^T

  const int T = 256;
  auto g1 = [&](long long tot) { return dim3((unsigned)((tot + T - 1) / T)); };

  // ---- degrees: deg = 1 + in-degree; dinv = rsqrt(deg)
  fill_f32<<<g1(N), T, 0, stream>>>(dinv, 1.0f, N);
  deg_accum<<<g1(E), T, 0, stream>>>(dst, dinv, E);
  rsqrt_inplace<<<g1(N), T, 0, stream>>>(dinv, N);

  const int Mblk = (N + BM - 1) / BM;

  // =============== layer 1: 126(->128) -> 512 ===============
  convert_x_bf16<<<g1((long long)N * 128), T, 0, stream>>>(x, xb, N, D_IN, 128);
  convert_w_bf16_t<<<g1((long long)512 * 128), T, 0, stream>>>(W1, wbt, D_IN, 512, 128, 512);
  gemm_bf16_wmma<<<dim3(Mblk, 512 / BN), 256, 0, stream>>>(xb, wbt, h, N, 128, HSTRIDE);
  self_bias<<<g1((long long)N * 512), T, 0, stream>>>(h, dinv, b1, agg, N, 512, HSTRIDE);
  scatter_edges<<<E, T, 0, stream>>>(h, agg, src, dst, dinv, E, 512, HSTRIDE);
  silu_to_bf16<<<g1((long long)N * 512), T, 0, stream>>>(agg, xb, N, 512);

  // =============== layer 2: 512 -> 2048 ===============
  convert_w_bf16_t<<<g1((long long)2048 * 512), T, 0, stream>>>(W2, wbt, 512, 2048, 512, 2048);
  gemm_bf16_wmma<<<dim3(Mblk, 2048 / BN), 256, 0, stream>>>(xb, wbt, h, N, 512, HSTRIDE);
  self_bias<<<g1((long long)N * 2048), T, 0, stream>>>(h, dinv, b2, agg, N, 2048, HSTRIDE);
  scatter_edges<<<E, T, 0, stream>>>(h, agg, src, dst, dinv, E, 2048, HSTRIDE);
  silu_to_bf16<<<g1((long long)N * 2048), T, 0, stream>>>(agg, xb, N, 2048);

  // =============== layer 3: 2048 -> 2000(->2048) ===============
  convert_w_bf16_t<<<g1((long long)2048 * 2048), T, 0, stream>>>(W3, wbt, 2048, 2000, 2048, 2048);
  gemm_bf16_wmma<<<dim3(Mblk, 2048 / BN), 256, 0, stream>>>(xb, wbt, h, N, 2048, HSTRIDE);
  // final output written directly into d_out (stride 2000, no activation)
  self_bias<<<g1((long long)N * 2000), T, 0, stream>>>(h, dinv, b3, out, N, 2000, 2000);
  scatter_edges<<<E, T, 0, stream>>>(h, out, src, dst, dinv, E, 2000, 2000);
}
